// HashSAT_2705829396953
// MI455X (gfx1250) — matrix operand order repaired and checked by reading, not compiled, not run on hardware
//
#include <hip/hip_runtime.h>
#include <math.h>

// ---------------------------------------------------------------------------
// Types for CDNA5 WMMA (wave32): V_WMMA_F32_16X16X32_BF16
// ---------------------------------------------------------------------------
typedef __attribute__((ext_vector_type(16))) __bf16 bf16x16;
typedef __attribute__((ext_vector_type(8)))  __bf16 bf16x8;
typedef __attribute__((ext_vector_type(8)))  float  f32x8;
typedef __attribute__((ext_vector_type(4)))  float  f32x4;

#define N_VARS  4096
#define N_LITS  8192
#define N_GATES 16384
#define TSTEPS  8
#define NORMV   0.125f  /* 1/sqrt(64) */
#define LDS_LDW 68      /* padded row stride (floats): 272B, 16B-aligned, no 2-way bank repeat */

__device__ __forceinline__ __bf16 f2bf(float x) { return (__bf16)x; }

__device__ __forceinline__ float sigmoidf_(float x) {
    return 1.0f / (1.0f + expf(-x));
}

// ---------------------------------------------------------------------------
// WMMA fragment loaders.  ISA layout (CDNA5 ISA §7.12.2), wave32:
//   A (16x32 bf16): lane L: m = L&15, half = L>>4.
//     elems 0..7  = A[m][k0 + 8*half + 0..7]
//     elems 8..15 = A[m][k0 + 16 + 8*half + 0..7]
//   B (32x16 bf16): lane L: n = L&15, half = L>>4.
//     elems 0..7 = B[k0 + 8*half + j][n], 8..15 = B[k0 + 16 + 8*half + j][n]   (we store B transposed: Bt[n][k])
//   C/D (16x16 f32): lane L: n = L&15; elem v = D[v + 8*(L>>4)][n]
// ---------------------------------------------------------------------------
__device__ __forceinline__ bf16x16 pack_a(f32x4 c0, f32x4 c1, f32x4 c2, f32x4 c3) {
    bf16x16 a;
#pragma unroll
    for (int j = 0; j < 4; ++j) {
        a[j]      = f2bf(c0[j]);  a[4 + j]  = f2bf(c1[j]);
        a[8 + j]  = f2bf(c2[j]);  a[12 + j] = f2bf(c3[j]);
    }
    return a;
}

__device__ __forceinline__ bf16x16 load_a_f32(const float* __restrict__ X, size_t ld,
                                              int m_base, int k0, int lane) {
    int half = lane >> 4, m = lane & 15;
    const float* row = X + (size_t)(m_base + m) * ld + k0 + half * 8;
    const f32x4* q = (const f32x4*)row;
    return pack_a(q[0], q[1], q[4], q[5]);
}

// A fragment from a wave-private LDS tile (row stride LDS_LDW floats)
__device__ __forceinline__ bf16x16 load_a_lds(const float* tile, int k0, int lane) {
    int half = lane >> 4, m = lane & 15;
    const float* row = tile + m * LDS_LDW + k0 + half * 8;
    const f32x4* q = (const f32x4*)row;
    return pack_a(q[0], q[1], q[4], q[5]);
}

__device__ __forceinline__ bf16x16 load_a_bf16(const __bf16* __restrict__ X, size_t ld,
                                               int m_base, int k0, int lane) {
    int half = lane >> 4, m = lane & 15;
    const __bf16* row = X + (size_t)(m_base + m) * ld + k0 + half * 8;
    union { bf16x16 v; bf16x8 h[2]; } u;
    u.h[0] = *(const bf16x8*)row;
    u.h[1] = *(const bf16x8*)(row + 16);
    return u.v;
}

// Bt is [N x K] row-major bf16 (i.e. B transposed), ld = K
__device__ __forceinline__ bf16x16 load_b(const __bf16* __restrict__ Bt, size_t ld,
                                          int n_base, int k0, int lane) {
    int half = lane >> 4, n = lane & 15;
    const __bf16* p = Bt + (size_t)(n_base + n) * ld + k0 + half * 8;
    union { bf16x16 v; bf16x8 h[2]; } u;
    u.h[0] = *(const bf16x8*)p;
    u.h[1] = *(const bf16x8*)(p + 16);
    return u.v;
}

#define WMMA_BF16(a, b, c) \
    __builtin_amdgcn_wmma_f32_16x16x32_bf16(false, (a), false, (b), (short)0, (c), false, false)

// ---------------------------------------------------------------------------
// Fused multi-layer MLP: NL layers of y = act(x @ W_l + b_l), ReLU between
// layers (and on last layer iff RELU_LAST).  One wave owns a 16-row strip for
// the whole depth; between layers the D-tile -> A-fragment relayout (a 16x16
// lane transpose) goes through a wave-private LDS tile (no barriers: same-wave
// DS ordering).  Weights transposed bf16 [l][64(n) x 64(k)], layer stride 4096.
// TBF16: final layer written transposed bf16 OUTt[64, M] (big-GEMM B panel).
// ---------------------------------------------------------------------------
template <int NL, bool TBF16, bool RELU_LAST>
__global__ __launch_bounds__(256) void mlp_kernel(
        const float* __restrict__ X, const __bf16* __restrict__ Wt,
        const float* __restrict__ bias,
        float* __restrict__ OUT, __bf16* __restrict__ OUTt, int M) {
    __shared__ float sbuf[8 * 16 * LDS_LDW];
    int lane = threadIdx.x & 31;
    int wave = threadIdx.x >> 5;
    int m0 = (blockIdx.x * 8 + wave) * 16;
    if (m0 >= M) return;
    float* tile = sbuf + wave * 16 * LDS_LDW;
    int half = lane >> 4, n = lane & 15;
#pragma unroll
    for (int l = 0; l < NL; ++l) {
        f32x8 acc[4] = {};
#pragma unroll
        for (int kt = 0; kt < 64; kt += 32) {
            bf16x16 a = (l == 0) ? load_a_f32(X, 64, m0, kt, lane)
                                 : load_a_lds(tile, kt, lane);
#pragma unroll
            for (int nt = 0; nt < 4; ++nt) {
                bf16x16 b = load_b(Wt + l * 4096, 64, nt * 16, kt, lane);
                acc[nt] = WMMA_BF16(a, b, acc[nt]);
            }
        }
        if (l < NL - 1) {
            // ReLU + park in LDS as next layer's input (lane transpose via LDS)
#pragma unroll
            for (int nt = 0; nt < 4; ++nt) {
                int ncol = nt * 16 + n;
                float bb = bias[l * 64 + ncol];
#pragma unroll
                for (int v = 0; v < 8; ++v)
                    tile[(8 * half + v) * LDS_LDW + ncol] = fmaxf(acc[nt][v] + bb, 0.0f);
            }
        } else {
#pragma unroll
            for (int nt = 0; nt < 4; ++nt) {
                int ncol = nt * 16 + n;
                float bb = bias[l * 64 + ncol];
                if constexpr (TBF16) {
                    __bf16 t[8];
#pragma unroll
                    for (int v = 0; v < 8; ++v) t[v] = f2bf(acc[nt][v] + bb);
                    *(bf16x8*)(OUTt + (size_t)ncol * M + m0 + 8 * half) = *(bf16x8*)t;
                } else {
#pragma unroll
                    for (int v = 0; v < 8; ++v) {
                        float x = acc[nt][v] + bb;
                        if constexpr (RELU_LAST) x = fmaxf(x, 0.0f);
                        OUT[(size_t)(m0 + 8 * half + v) * 64 + ncol] = x;
                    }
                }
            }
        }
    }
}

// ---------------------------------------------------------------------------
// Big GEMM: OUT[M,64] += Abig[M,K] @ Bt[64,K]^T.
// 2 row-strips (32 rows) per wave: each B fragment feeds two WMMAs.
// K split over gridDim.y, accumulated with f32 atomics (OUT zeroed first).
// ABF: A already bf16; else convert fp32 A in-register (workspace fallback).
// ---------------------------------------------------------------------------
template <bool ABF>
__global__ __launch_bounds__(256) void big_gemm_kernel(
        const __bf16* __restrict__ Abf, const float* __restrict__ Af,
        const __bf16* __restrict__ Bt, float* __restrict__ OUT,
        int M, int K) {
    int lane = threadIdx.x & 31;
    int wave = threadIdx.x >> 5;
    int m0 = (blockIdx.x * 8 + wave) * 32;
    if (m0 >= M) return;
    int kc = K / gridDim.y;
    int kb = blockIdx.y * kc, ke = kb + kc;
    f32x8 acc[2][4] = {};
#pragma unroll 2
    for (int k = kb; k < ke; k += 32) {
        bf16x16 a0, a1;
        if constexpr (ABF) {
            a0 = load_a_bf16(Abf, K, m0, k, lane);
            a1 = load_a_bf16(Abf, K, m0 + 16, k, lane);
        } else {
            a0 = load_a_f32(Af, K, m0, k, lane);
            a1 = load_a_f32(Af, K, m0 + 16, k, lane);
        }
#pragma unroll
        for (int nt = 0; nt < 4; ++nt) {
            bf16x16 b = load_b(Bt, K, nt * 16, k, lane);
            acc[0][nt] = WMMA_BF16(a0, b, acc[0][nt]);
            acc[1][nt] = WMMA_BF16(a1, b, acc[1][nt]);
        }
    }
    int half = lane >> 4, n = lane & 15;
#pragma unroll
    for (int r = 0; r < 2; ++r)
#pragma unroll
        for (int nt = 0; nt < 4; ++nt)
#pragma unroll
            for (int v = 0; v < 8; ++v)
                atomicAdd(&OUT[(size_t)(m0 + r * 16 + 8 * half + v) * 64 + nt * 16 + n],
                          acc[r][nt][v]);
}

// ---------------------------------------------------------------------------
// Fused LSTM cell (C side): gates = X@WihT + H@WhhT + b ; then elementwise.
// Column-split over gridDim.y (4 groups, one 16-col n'-tile each) -> only
// 4 accumulators: tiles {q0*16, 64+q0*16, 128+.., 192+..} = i/f/g/o for this
// wave's columns, lane-local.  H ping-ponged (other waves read all H columns
// for the GEMM); C updated in place (column-disjoint).
// Wih/Whh stored bf16 row-major [256(n) x 64(k)] == already the B-transpose.
// ---------------------------------------------------------------------------
__global__ __launch_bounds__(256) void lstm_c_kernel(
        const float* __restrict__ X, const __bf16* __restrict__ Wih,
        const __bf16* __restrict__ Whh, const float* __restrict__ bias,
        const float* __restrict__ Hin, float* __restrict__ Hout,
        float* __restrict__ C, int M) {
    int lane = threadIdx.x & 31;
    int wave = threadIdx.x >> 5;
    int m0 = (blockIdx.x * 8 + wave) * 16;
    if (m0 >= M) return;
    int q0 = blockIdx.y;      // n'-tile (0..3) of this column group
    f32x8 acc[4] = {};        // acc[g] = gate g for columns q0*16..q0*16+15
#pragma unroll
    for (int kt = 0; kt < 64; kt += 32) {
        bf16x16 ax = load_a_f32(X, 64, m0, kt, lane);
        bf16x16 ah = load_a_f32(Hin, 64, m0, kt, lane);
#pragma unroll
        for (int g = 0; g < 4; ++g) {
            int nb = g * 64 + q0 * 16;
            acc[g] = WMMA_BF16(ax, load_b(Wih, 64, nb, kt, lane), acc[g]);
            acc[g] = WMMA_BF16(ah, load_b(Whh, 64, nb, kt, lane), acc[g]);
        }
    }
    int half = lane >> 4, n = lane & 15;
    int ncol = q0 * 16 + n;
    float bi = bias[ncol], bf_ = bias[64 + ncol], bg = bias[128 + ncol], bo = bias[192 + ncol];
#pragma unroll
    for (int v = 0; v < 8; ++v) {
        size_t m = (size_t)(m0 + 8 * half + v);
        float ig = sigmoidf_(acc[0][v] + bi);
        float fg = sigmoidf_(acc[1][v] + bf_);
        float gg = tanhf(acc[2][v] + bg);
        float og = sigmoidf_(acc[3][v] + bo);
        float c2 = fg * C[m * 64 + ncol] + ig * gg;
        C[m * 64 + ncol] = c2;
        Hout[m * 64 + ncol] = og * tanhf(c2);
    }
}

// ---------------------------------------------------------------------------
// Fused LSTM cell (L side): gates = Cg@WihL[:, :64]^T + flip(Lst)@WihL[:, 64:]^T
//                                   + H@WhhL^T + b.  flip(row m) = row m^4096.
// Same column-split (x4) as lstm_c; L_state and hidden both ping-ponged.
// ---------------------------------------------------------------------------
__global__ __launch_bounds__(256) void lstm_l_kernel(
        const float* __restrict__ Cg, const float* __restrict__ Lin,
        const __bf16* __restrict__ WihL, const __bf16* __restrict__ WhhL,
        const float* __restrict__ bias,
        const float* __restrict__ Hin, float* __restrict__ Hout,
        float* __restrict__ Lout, int M) {
    int lane = threadIdx.x & 31;
    int wave = threadIdx.x >> 5;
    int m0 = (blockIdx.x * 8 + wave) * 16;
    if (m0 >= M) return;
    int m0f = m0 ^ N_VARS;  // flipped strip (16 | 4096 so strips map cleanly)
    int q0 = blockIdx.y;
    f32x8 acc[4] = {};
#pragma unroll
    for (int kt = 0; kt < 64; kt += 32) {
        bf16x16 a1 = load_a_f32(Cg, 64, m0, kt, lane);
        bf16x16 a2 = load_a_f32(Lin, 64, m0f, kt, lane);
        bf16x16 a3 = load_a_f32(Hin, 64, m0, kt, lane);
#pragma unroll
        for (int g = 0; g < 4; ++g) {
            int nb = g * 64 + q0 * 16;
            f32x8 c = acc[g];
            c = WMMA_BF16(a1, load_b(WihL, 128, nb, kt, lane), c);
            c = WMMA_BF16(a2, load_b(WihL + 64, 128, nb, kt, lane), c);
            c = WMMA_BF16(a3, load_b(WhhL, 64, nb, kt, lane), c);
            acc[g] = c;
        }
    }
    int half = lane >> 4, n = lane & 15;
    int ncol = q0 * 16 + n;
    float bi = bias[ncol], bf_ = bias[64 + ncol], bg = bias[128 + ncol], bo = bias[192 + ncol];
#pragma unroll
    for (int v = 0; v < 8; ++v) {
        size_t m = (size_t)(m0 + 8 * half + v);
        float ig = sigmoidf_(acc[0][v] + bi);
        float fg = sigmoidf_(acc[1][v] + bf_);
        float gg = tanhf(acc[2][v] + bg);
        float og = sigmoidf_(acc[3][v] + bo);
        float c2 = fg * Lin[m * 64 + ncol] + ig * gg;
        Lout[m * 64 + ncol] = c2;
        Hout[m * 64 + ncol] = og * tanhf(c2);
    }
}

// ---------------------------------------------------------------------------
// Setup / utility kernels
// ---------------------------------------------------------------------------
__global__ void init_states_kernel(const float* __restrict__ Li, const float* __restrict__ Ci,
                                   float* __restrict__ Lst, float* __restrict__ Lh,
                                   float* __restrict__ Cst, float* __restrict__ Ch) {
    size_t i = (size_t)blockIdx.x * blockDim.x + threadIdx.x;  // over 16384*64
    int d = (int)(i & 63);
    size_t row = i >> 6;
    if (row < N_GATES) { Cst[i] = Ci[d] * NORMV; Ch[i] = 0.0f; }
    if (row < N_LITS)  { Lst[i] = Li[d] * NORMV; Lh[i] = 0.0f; }
}

__global__ void apply_obs_kernel(const int* __restrict__ idx, const float* __restrict__ val,
                                 float* __restrict__ Lst) {
    int t = blockIdx.x * blockDim.x + threadIdx.x;  // 256 obs * 64
    int o = t >> 6, d = t & 63;
    if (o < 256) {
        int r = idx[o];
        Lst[(size_t)r * 64 + d] = val[o];
        Lst[(size_t)(r + N_VARS) * 64 + d] = 1.0f - val[o];
    }
}

__global__ void cvt_f32_bf16_kernel(const float* __restrict__ s, __bf16* __restrict__ d, size_t n8) {
    size_t i = (size_t)blockIdx.x * blockDim.x + threadIdx.x;
    size_t stride = (size_t)gridDim.x * blockDim.x;
    for (; i < n8; i += stride) {
        const f32x4* q = (const f32x4*)(s + i * 8);
        f32x4 a = q[0], b = q[1];
        __bf16 t[8];
#pragma unroll
        for (int j = 0; j < 4; ++j) { t[j] = f2bf(a[j]); t[4 + j] = f2bf(b[j]); }
        *(bf16x8*)(d + i * 8) = *(bf16x8*)t;
    }
}

// Wt[l][n][k] = W[l][k][n]  (64x64 layers)
__global__ void w64_transpose_bf16_kernel(const float* __restrict__ s, __bf16* __restrict__ d,
                                          int layers) {
    int i = blockIdx.x * blockDim.x + threadIdx.x;
    if (i < layers * 4096) {
        int l = i >> 12, r = i & 4095, n = r >> 6, k = r & 63;
        d[(l << 12) + (n << 6) + k] = f2bf(s[(l << 12) + (k << 6) + n]);
    }
}

__global__ void zero_f32_kernel(float* __restrict__ p, size_t n) {
    size_t i = (size_t)blockIdx.x * blockDim.x + threadIdx.x;
    size_t stride = (size_t)gridDim.x * blockDim.x;
    for (; i < n; i += stride) p[i] = 0.0f;
}

__global__ void vote_kernel(const float* __restrict__ v1, const float* __restrict__ Wvo,
                            float* __restrict__ out) {
    int t = blockIdx.x * blockDim.x + threadIdx.x;
    if (t < N_VARS) {
        float s1 = 0.0f, s2 = 0.0f;
#pragma unroll 8
        for (int k = 0; k < 64; ++k) {
            float w = Wvo[k];
            s1 += v1[(size_t)t * 64 + k] * w;
            s2 += v1[(size_t)(t + N_VARS) * 64 + k] * w;
        }
        out[t] = sigmoidf_(s1 - s2);  // bvo cancels in the difference
    }
}

// ---------------------------------------------------------------------------
// Host orchestration
// ---------------------------------------------------------------------------
extern "C" void kernel_launch(void* const* d_in, const int* in_sizes, int n_in,
                              void* d_out, int out_size, void* d_ws, size_t ws_size,
                              hipStream_t stream) {
    (void)in_sizes; (void)n_in; (void)out_size;
    const float* A    = (const float*)d_in[0];
    const float* AT   = (const float*)d_in[1];
    const int*   oix  = (const int*)d_in[2];
    const float* oval = (const float*)d_in[3];
    const float* Li   = (const float*)d_in[4];
    const float* Ci   = (const float*)d_in[5];
    const float* WmL  = (const float*)d_in[6];
    const float* bmL  = (const float*)d_in[7];
    const float* WmC  = (const float*)d_in[8];
    const float* bmC  = (const float*)d_in[9];
    const float* Wv   = (const float*)d_in[10];
    const float* bv   = (const float*)d_in[11];
    const float* Wvo  = (const float*)d_in[12];
    const float* WihL = (const float*)d_in[14];
    const float* WhhL = (const float*)d_in[15];
    const float* bL   = (const float*)d_in[16];
    const float* WihC = (const float*)d_in[17];
    const float* WhhC = (const float*)d_in[18];
    const float* bC   = (const float*)d_in[19];

    char* base = (char*)d_ws;
    size_t off = 0;
    auto carve = [&](size_t b) -> void* {
        void* r = base + off;
        off = (off + b + 255) & ~(size_t)255;
        return r;
    };
    float*  Lst0  = (float*)carve((size_t)N_LITS * 64 * 4);
    float*  Lst1  = (float*)carve((size_t)N_LITS * 64 * 4);
    float*  Lh0   = (float*)carve((size_t)N_LITS * 64 * 4);
    float*  Lh1   = (float*)carve((size_t)N_LITS * 64 * 4);
    float*  Cst   = (float*)carve((size_t)N_GATES * 64 * 4);
    float*  Ch0   = (float*)carve((size_t)N_GATES * 64 * 4);
    float*  Ch1   = (float*)carve((size_t)N_GATES * 64 * 4);
    float*  Msg   = (float*)carve((size_t)N_GATES * 64 * 4);   // L_msg, reused for C_msg
    __bf16* Bt    = (__bf16*)carve((size_t)64 * N_GATES * 2);  // transposed mlp output
    float*  tA    = (float*)carve((size_t)N_LITS * 64 * 4);    // vote-head intermediate
    __bf16* WmLt  = (__bf16*)carve(3 * 4096 * 2);
    __bf16* WmCt  = (__bf16*)carve(3 * 4096 * 2);
    __bf16* Wvt   = (__bf16*)carve(2 * 4096 * 2);
    __bf16* WihLb = (__bf16*)carve(256 * 128 * 2);
    __bf16* WhhLb = (__bf16*)carve(256 * 64 * 2);
    __bf16* WihCb = (__bf16*)carve(256 * 64 * 2);
    __bf16* WhhCb = (__bf16*)carve(256 * 64 * 2);
    const size_t ASZ = (size_t)N_LITS * N_GATES;  // 128M elements
    __bf16* Abf  = (__bf16*)carve(ASZ * 2);
    __bf16* ATbf = (__bf16*)carve(ASZ * 2);
    bool bigOK = (off <= ws_size);

    // ---- one-time conversions (re-done every call; deterministic) ----
    w64_transpose_bf16_kernel<<<(3 * 4096 + 255) / 256, 256, 0, stream>>>(WmL, WmLt, 3);
    w64_transpose_bf16_kernel<<<(3 * 4096 + 255) / 256, 256, 0, stream>>>(WmC, WmCt, 3);
    w64_transpose_bf16_kernel<<<(2 * 4096 + 255) / 256, 256, 0, stream>>>(Wv, Wvt, 2);
    cvt_f32_bf16_kernel<<<16, 256, 0, stream>>>(WihL, WihLb, (256 * 128) / 8);
    cvt_f32_bf16_kernel<<<8, 256, 0, stream>>>(WhhL, WhhLb, (256 * 64) / 8);
    cvt_f32_bf16_kernel<<<8, 256, 0, stream>>>(WihC, WihCb, (256 * 64) / 8);
    cvt_f32_bf16_kernel<<<8, 256, 0, stream>>>(WhhC, WhhCb, (256 * 64) / 8);
    if (bigOK) {
        cvt_f32_bf16_kernel<<<8192, 256, 0, stream>>>(A, Abf, ASZ / 8);
        cvt_f32_bf16_kernel<<<8192, 256, 0, stream>>>(AT, ATbf, ASZ / 8);
    }

    init_states_kernel<<<(N_GATES * 64) / 256, 256, 0, stream>>>(Li, Ci, Lst0, Lh0, Cst, Ch0);
    apply_obs_kernel<<<64, 256, 0, stream>>>(oix, oval, Lst0);

    float* Lcur = Lst0; float* Lnxt = Lst1;
    float* LhCur = Lh0; float* LhNxt = Lh1;
    float* ChCur = Ch0; float* ChNxt = Ch1;
    for (int t = 0; t < TSTEPS; ++t) {
        // ---- L-side fused 3-layer MLP -> transposed bf16 panel Bt [64, 8192] ----
        mlp_kernel<3, true, false><<<64, 256, 0, stream>>>(Lcur, WmLt, bmL, nullptr, Bt, N_LITS);
        // ---- L_msg = A^T @ mlp  (16384 x 64, K = 8192) ----
        zero_f32_kernel<<<1024, 256, 0, stream>>>(Msg, (size_t)N_GATES * 64);
        if (bigOK)
            big_gemm_kernel<true><<<dim3(N_GATES / 256, 4), 256, 0, stream>>>(ATbf, nullptr, Bt, Msg, N_GATES, N_LITS);
        else
            big_gemm_kernel<false><<<dim3(N_GATES / 256, 4), 256, 0, stream>>>(nullptr, AT, Bt, Msg, N_GATES, N_LITS);
        // ---- C LSTM (fused gates + cell), column-split x4, hidden ping-pong ----
        lstm_c_kernel<<<dim3(N_GATES / 128, 4), 256, 0, stream>>>(Msg, WihCb, WhhCb, bC,
                                                                  ChCur, ChNxt, Cst, N_GATES);
        { float* tmp = ChCur; ChCur = ChNxt; ChNxt = tmp; }
        // ---- C-side fused 3-layer MLP -> Bt [64, 16384] ----
        mlp_kernel<3, true, false><<<128, 256, 0, stream>>>(Cst, WmCt, bmC, nullptr, Bt, N_GATES);
        // ---- C_msg(gemm part) = A @ mlp  (8192 x 64, K = 16384) ----
        zero_f32_kernel<<<512, 256, 0, stream>>>(Msg, (size_t)N_LITS * 64);
        if (bigOK)
            big_gemm_kernel<true><<<dim3(N_LITS / 256, 8), 256, 0, stream>>>(Abf, nullptr, Bt, Msg, N_LITS, N_GATES);
        else
            big_gemm_kernel<false><<<dim3(N_LITS / 256, 8), 256, 0, stream>>>(nullptr, A, Bt, Msg, N_LITS, N_GATES);
        // ---- L LSTM (fused 3-term gates + cell, flipped concat, ping-pong) ----
        lstm_l_kernel<<<dim3(N_LITS / 128, 4), 256, 0, stream>>>(Msg, Lcur, WihLb, WhhLb, bL,
                                                                 LhCur, LhNxt, Lnxt, N_LITS);
        { float* tmp = Lcur; Lcur = Lnxt; Lnxt = tmp; }
        { float* tmp = LhCur; LhCur = LhNxt; LhNxt = tmp; }
    }

    // ---- vote head: fused 2-layer ReLU MLP, then per-var vote reduction ----
    mlp_kernel<2, false, true><<<64, 256, 0, stream>>>(Lcur, Wvt, bv, tA, nullptr, N_LITS);
    vote_kernel<<<N_VARS / 256, 256, 0, stream>>>(tA, Wvo, (float*)d_out);
}
